// MlpPartlyIndepNet_79826262163666
// MI455X (gfx1250) — matrix operand order repaired and checked by reading, not compiled, not run on hardware
//
#include <hip/hip_runtime.h>
#include <hip/hip_bf16.h>
#include <math.h>

// ---------------------------------------------------------------------------
// MI455X fused MLP (trunk + 11 readout heads), bf16 WMMA, BN folded, eval mode
// ---------------------------------------------------------------------------

typedef __bf16 v16bf __attribute__((ext_vector_type(16)));
typedef __bf16 v8bf  __attribute__((ext_vector_type(8)));
typedef float  v8f   __attribute__((ext_vector_type(8)));

#define MT      64            // rows (batch) per workgroup
#define MTILES  (MT / 16)     // 4 WMMA M-tiles
#define NWAVES  8
#define THREADS (NWAVES * 32)

__device__ __forceinline__ float elu1(float z) {
    return z > 0.f ? z : (__expf(z) - 1.f);
}

// ---------------------------------------------------------------------------
// Prep: fold BN (scale s[n]=g/sqrt(v+eps), bias (b-m)*s+be) into the weight,
// convert to bf16, and pre-swizzle into WMMA B-operand tile layout:
//   tile(nt,kt): element idx = lane*16+e holds W'[n][k] with
//   n = nt*16 + (lane&15), k = kt*32 + (lane>=16 ? 16 : 0) + e
// Padded n/k positions are zero (so padded outputs become ELU(0)=0).
// ---------------------------------------------------------------------------
__global__ void fold_pack(const float* __restrict__ W, const float* __restrict__ b,
                          const float* __restrict__ g, const float* __restrict__ be,
                          const float* __restrict__ m, const float* __restrict__ v,
                          __hip_bfloat16* __restrict__ Wp, float* __restrict__ bp,
                          int Nreal, int Kreal, int ntiles, int ktiles)
{
    const int idx   = blockIdx.x * blockDim.x + threadIdx.x;
    const int total = ntiles * ktiles * 512;
    if (idx < total) {
        const int tile   = idx >> 9;
        const int within = idx & 511;
        const int lane   = within >> 4;
        const int e      = within & 15;
        const int nt     = tile / ktiles;
        const int kt     = tile - nt * ktiles;
        const int n      = nt * 16 + (lane & 15);
        const int k      = kt * 32 + ((lane >> 4) << 4) + e;
        float val = 0.f;
        if (n < Nreal && k < Kreal) {
            const float s = g[n] * rsqrtf(v[n] + 1e-5f);
            val = W[(size_t)n * Kreal + k] * s;
        }
        Wp[idx] = __float2bfloat16(val);
    }
    if (idx < ntiles * 16) {
        const int n = idx;
        float bv = 0.f;
        if (n < Nreal) {
            const float s = g[n] * rsqrtf(v[n] + 1e-5f);
            bv = (b[n] - m[n]) * s + be[n];
        }
        bp[idx] = bv;
    }
}

// ---------------------------------------------------------------------------
// One fused layer: LDS bf16 activations (M=64 rows) x packed bf16 weights ->
// bias + ELU -> LDS bf16. Output tiles round-robined over the 8 waves.
// A-operand (matches packed B): lanes 0-15 take K {0..15},
// lanes 16-31 take K {16..31} (two 16B LDS loads + shuffle).
// ---------------------------------------------------------------------------
template<int NT, int KT, int LDIN, int LDOUT>
__device__ __forceinline__ void gemm_layer(const __hip_bfloat16* __restrict__ actIn,
                                           __hip_bfloat16* __restrict__ actOut,
                                           const __hip_bfloat16* __restrict__ Wp,
                                           const float* __restrict__ bias,
                                           int wave, int lane)
{
    const int hi  = lane >> 4;        // 0 or 1
    const int l15 = lane & 15;
    for (int t = wave; t < MTILES * NT; t += NWAVES) {
        const int mt = t / NT;
        const int nt = t - mt * NT;
        v8f acc = {0.f, 0.f, 0.f, 0.f, 0.f, 0.f, 0.f, 0.f};
        const __hip_bfloat16* wp = Wp + (size_t)nt * KT * 512 + lane * 16;
        const __hip_bfloat16* ar = actIn + (mt * 16 + l15) * LDIN + hi * 16;
#pragma unroll
        for (int kt = 0; kt < KT; ++kt) {
            const __hip_bfloat16* ap = ar + kt * 32;
            v8bf alo = *(const v8bf*)(ap);          // K: base + {0..7}
            v8bf ahi = *(const v8bf*)(ap + 8);      // K: base + {8..15}
            v16bf a  = __builtin_shufflevector(alo, ahi,
                        0,1,2,3,4,5,6,7,8,9,10,11,12,13,14,15);
            v16bf b  = *(const v16bf*)(wp + kt * 512);
            acc = __builtin_amdgcn_wmma_f32_16x16x32_bf16(
                      false, a, false, b, (short)0, acc, false, false);
        }
        const int   ncol  = nt * 16 + l15;
        const float bv    = bias[ncol];
        const int   rbase = mt * 16 + hi * 8;   // C/D: lanes>=16 hold M = 8+r
#pragma unroll
        for (int r = 0; r < 8; ++r) {
            float z = acc[r] + bv;
            actOut[(rbase + r) * LDOUT + ncol] = __float2bfloat16(elu1(z));
        }
    }
}

// ---------------------------------------------------------------------------
// Fully fused forward: x0 tile -> trunk (3 layers) -> +x1 -> 11 heads -> out
// All intermediates stay in LDS; HBM traffic = x0 + out + L2-resident weights.
// ---------------------------------------------------------------------------
__global__ __launch_bounds__(THREADS) void fused_mlp(
    const float* __restrict__ x0, const float* __restrict__ x1,
    const __hip_bfloat16* __restrict__ Wt,  const float* __restrict__ bt,
    const __hip_bfloat16* __restrict__ Wh1, const float* __restrict__ bh1,
    const __hip_bfloat16* __restrict__ Wh2, const float* __restrict__ bh2,
    const __hip_bfloat16* __restrict__ Wh3, const float* __restrict__ bh3,
    const float* __restrict__ Wf, const float* __restrict__ bfin,
    float* __restrict__ out)
{
    __shared__ __align__(32) __hip_bfloat16 A [MT][224];
    __shared__ __align__(32) __hip_bfloat16 Bh[MT][224];
    __shared__ __align__(32) __hip_bfloat16 Y1[MT][128];
    __shared__ __align__(32) __hip_bfloat16 Y2[MT][64];
    __shared__ __align__(32) __hip_bfloat16 Y3[MT][32];

    const int tid  = threadIdx.x;
    const int wave = tid >> 5;
    const int lane = tid & 31;
    const int row0 = blockIdx.x * MT;

    constexpr int TP = 14 * 7 * 512;

    // Warm the cache hierarchy with the trunk L1 weights while x0 streams in
    // (lowered to global_prefetch_b8; one byte per 128B line).
    for (int i = tid * 64; i < TP; i += THREADS * 64)
        __builtin_prefetch(Wt + i, 0, 1);

    // Load the x0 tile (f32 -> bf16), 16B vector global loads.
    for (int i = tid; i < MT * 224 / 4; i += THREADS) {
        const int r = i / 56;
        const int c = (i - r * 56) * 4;
        const float4 f = *(const float4*)(x0 + (size_t)(row0 + r) * 224 + c);
        A[r][c + 0] = __float2bfloat16(f.x);
        A[r][c + 1] = __float2bfloat16(f.y);
        A[r][c + 2] = __float2bfloat16(f.z);
        A[r][c + 3] = __float2bfloat16(f.w);
    }
    __syncthreads();

    gemm_layer<14,7,224,224>(&A[0][0],  &Bh[0][0], Wt + 0 * TP, bt + 0 * 224, wave, lane);
    __syncthreads();
    gemm_layer<14,7,224,224>(&Bh[0][0], &A[0][0],  Wt + 1 * TP, bt + 1 * 224, wave, lane);
    __syncthreads();
    gemm_layer<14,7,224,224>(&A[0][0],  &Bh[0][0], Wt + 2 * TP, bt + 2 * 224, wave, lane);
    __syncthreads();
    // Splice auxiliary scalar feature into column 200 (cols 201..223 are 0).
    if (tid < MT) Bh[tid][200] = __float2bfloat16(x1[row0 + tid]);
    __syncthreads();

    for (int h = 0; h < 11; ++h) {
        gemm_layer<8,7,224,128>(&Bh[0][0], &Y1[0][0],
                                Wh1 + (size_t)h * (8 * 7 * 512), bh1 + h * 128, wave, lane);
        __syncthreads();
        gemm_layer<4,4,128,64>(&Y1[0][0], &Y2[0][0],
                               Wh2 + (size_t)h * (4 * 4 * 512), bh2 + h * 64, wave, lane);
        __syncthreads();
        gemm_layer<2,2,64,32>(&Y2[0][0], &Y3[0][0],
                              Wh3 + (size_t)h * (2 * 2 * 512), bh3 + h * 32, wave, lane);
        __syncthreads();
        if (tid < MT) {  // final 25 -> 1 dot per row (tiny; VALU)
            float s = bfin[h];
            const float* wf = Wf + h * 25;
#pragma unroll
            for (int i = 0; i < 25; ++i)
                s += __bfloat162float(Y3[tid][i]) * wf[i];
            out[(size_t)(row0 + tid) * 11 + h] = s;
        }
        __syncthreads();
    }
}

// ---------------------------------------------------------------------------
// Host launch
// ---------------------------------------------------------------------------
extern "C" void kernel_launch(void* const* d_in, const int* in_sizes, int n_in,
                              void* d_out, int out_size, void* d_ws, size_t ws_size,
                              hipStream_t stream)
{
    (void)in_sizes; (void)n_in; (void)out_size; (void)ws_size;
    // dict order: x0, x1, Wc[3], bc[3], gc[3], bec[3], mc[3], vc[3],
    //             hW[3], hb[3], hg[3], hbe[3], hm[3], hv[3], Wf, bf
    const float* x0 = (const float*)d_in[0];
    const float* x1 = (const float*)d_in[1];
    const float* Wc [3] = {(const float*)d_in[2],  (const float*)d_in[3],  (const float*)d_in[4]};
    const float* bc [3] = {(const float*)d_in[5],  (const float*)d_in[6],  (const float*)d_in[7]};
    const float* gc [3] = {(const float*)d_in[8],  (const float*)d_in[9],  (const float*)d_in[10]};
    const float* bec[3] = {(const float*)d_in[11], (const float*)d_in[12], (const float*)d_in[13]};
    const float* mc [3] = {(const float*)d_in[14], (const float*)d_in[15], (const float*)d_in[16]};
    const float* vc [3] = {(const float*)d_in[17], (const float*)d_in[18], (const float*)d_in[19]};
    const float* hW [3] = {(const float*)d_in[20], (const float*)d_in[21], (const float*)d_in[22]};
    const float* hb [3] = {(const float*)d_in[23], (const float*)d_in[24], (const float*)d_in[25]};
    const float* hg [3] = {(const float*)d_in[26], (const float*)d_in[27], (const float*)d_in[28]};
    const float* hbe[3] = {(const float*)d_in[29], (const float*)d_in[30], (const float*)d_in[31]};
    const float* hm [3] = {(const float*)d_in[32], (const float*)d_in[33], (const float*)d_in[34]};
    const float* hv [3] = {(const float*)d_in[35], (const float*)d_in[36], (const float*)d_in[37]};
    const float* Wf  = (const float*)d_in[38];
    const float* bff = (const float*)d_in[39];

    // Carve packed-weight workspace (~1.2 MB) out of d_ws.
    char* ws = (char*)d_ws;
    size_t off = 0;
    auto carve = [&](size_t bytes) -> void* {
        off = (off + 255) & ~(size_t)255;
        void* p = ws + off;
        off += bytes;
        return p;
    };
    constexpr int TP  = 14 * 7 * 512;
    constexpr int HP1 = 8 * 7 * 512;
    constexpr int HP2 = 4 * 4 * 512;
    constexpr int HP3 = 2 * 2 * 512;
    __hip_bfloat16* Wt  = (__hip_bfloat16*)carve((size_t)3  * TP  * 2);
    float*          bt  = (float*)         carve((size_t)3  * 224 * 4);
    __hip_bfloat16* Wh1 = (__hip_bfloat16*)carve((size_t)11 * HP1 * 2);
    float*          bh1 = (float*)         carve((size_t)11 * 128 * 4);
    __hip_bfloat16* Wh2 = (__hip_bfloat16*)carve((size_t)11 * HP2 * 2);
    float*          bh2 = (float*)         carve((size_t)11 * 64  * 4);
    __hip_bfloat16* Wh3 = (__hip_bfloat16*)carve((size_t)11 * HP3 * 2);
    float*          bh3 = (float*)         carve((size_t)11 * 32  * 4);

    // Fold BN + pack trunk weights (layer0 K=224, layers 1/2 K=200, N=200).
    for (int i = 0; i < 3; ++i) {
        const int K = (i == 0) ? 224 : 200;
        fold_pack<<<(TP + 255) / 256, 256, 0, stream>>>(
            Wc[i], bc[i], gc[i], bec[i], mc[i], vc[i],
            Wt + (size_t)i * TP, bt + i * 224, 200, K, 14, 7);
    }
    // Fold + pack the 11 heads, per layer.
    const int Nh[3]  = {100, 50, 25};
    const int Kh[3]  = {201, 100, 50};
    const int nth[3] = {8, 4, 2};
    const int kth[3] = {7, 4, 2};
    const int HP [3] = {HP1, HP2, HP3};
    const int bld[3] = {128, 64, 32};
    __hip_bfloat16* Whp[3] = {Wh1, Wh2, Wh3};
    float*          bhp[3] = {bh1, bh2, bh3};
    for (int l = 0; l < 3; ++l)
        for (int h = 0; h < 11; ++h)
            fold_pack<<<(HP[l] + 255) / 256, 256, 0, stream>>>(
                hW[l] + (size_t)h * Nh[l] * Kh[l], hb[l] + h * Nh[l],
                hg[l] + h * Nh[l], hbe[l] + h * Nh[l],
                hm[l] + h * Nh[l], hv[l] + h * Nh[l],
                Whp[l] + (size_t)h * HP[l], bhp[l] + h * bld[l],
                Nh[l], Kh[l], nth[l], kth[l]);

    fused_mlp<<<65536 / MT, THREADS, 0, stream>>>(
        x0, x1, Wt, bt, Wh1, bh1, Wh2, bh2, Wh3, bh3, Wf, bff, (float*)d_out);
}
